// ProteinEncoder_63230508531713
// MI455X (gfx1250) — compile-verified, matrix operand order checked by reading, not run on hardware
//
#include <hip/hip_runtime.h>

#define LNUM 4
#define DM   512
#define HN   8
#define DHD  64
#define FF   2048
#define EN   8
#define BB   4
#define SS   2048
#define TT   (BB*SS)          // 8192 tokens
#define SLOTCAP (2*TT + 256)  // routed slots incl. per-expert 16-padding

typedef __attribute__((ext_vector_type(8)))  _Float16 v8h;
typedef __attribute__((ext_vector_type(16))) _Float16 v16h;
typedef __attribute__((ext_vector_type(8)))  float    v8f;

// ---------------- WMMA helpers (CDNA5 gfx1250, wave32) ----------------
__device__ __forceinline__ v16h frag16(const _Float16* p) {
  v8h lo = *(const v8h*)(p);
  v8h hi = *(const v8h*)(p + 16);
  v16h a;
#pragma unroll
  for (int i = 0; i < 8; ++i) { a[i] = lo[i]; a[i + 8] = hi[i]; }
  return a;
}

__device__ __forceinline__ v8f wmma_f16(v16h a, v16h b, v8f c) {
  return __builtin_amdgcn_wmma_f32_16x16x32_f16(false, a, false, b, (short)0, c, false, false);
}

// One wave computes a 16(M) x 64(N) f32 tile. aRow = A + row(lane)*lda + khalf,
// bCol = BT + (n0+ln)*ldb + khalf, khalf = 8*(lane>=16).
__device__ __forceinline__ void gemm_16x64(const _Float16* aRow, const _Float16* bCol,
                                           int K, int ldb, v8f acc[4]) {
  for (int k = 0; k < K; k += 32) {
    v16h a = frag16(aRow + k);
#pragma unroll
    for (int s = 0; s < 4; ++s) {
      v16h b = frag16(bCol + (size_t)s * 16 * ldb + k);
      acc[s] = wmma_f16(a, b, acc[s]);
    }
  }
}

// ---------------- embedding + positional encoding ----------------
__global__ void k_embed(const int* __restrict__ tok, const float* __restrict__ emb,
                        float* __restrict__ x, _Float16* __restrict__ xh) {
  int idx = blockIdx.x * 256 + threadIdx.x;
  if (idx >= TT * DM) return;
  int t = idx >> 9, d = idx & 511;
  int pos = t & (SS - 1);
  float v = emb[(size_t)tok[t] * DM + d] * 22.62741699796952f; // sqrt(512)
  float expo = (float)(2 * (d >> 1)) * (1.0f / 512.0f);
  float ang = (float)pos * __expf(-9.210340371976184f * expo); // 10000^-expo
  v += (d & 1) ? __cosf(ang) : __sinf(ang);
  x[idx] = v;
  xh[idx] = (_Float16)v;
}

// transpose-convert f32 [batch,R,C] -> f16 [batch,C,R]
__global__ void k_convT(const float* __restrict__ src, _Float16* __restrict__ dst,
                        int R, int C, int total) {
  int idx = blockIdx.x * 256 + threadIdx.x;
  if (idx >= total) return;
  int rc = R * C;
  int e = idx / rc, i = idx - e * rc;
  int r = i / C, c = i - r * C;
  dst[(size_t)e * rc + (size_t)c * R + r] = (_Float16)src[idx];
}

// ---------------- QKV projection (writes f16 in attention layouts) ----------------
// mode 0: dst[b][h][s][dh]   (q and k)    mode 1: dst[b][h][dh][s]  (v transposed)
__global__ __launch_bounds__(32) void k_gemm_qkv(const _Float16* __restrict__ xh,
                                                 const _Float16* __restrict__ WT,
                                                 const float* __restrict__ bias,
                                                 _Float16* __restrict__ dst, int mode) {
  int lane = threadIdx.x, ln = lane & 15, hl = lane >> 4, khalf = 8 * hl;
  int t0 = blockIdx.x * 16, n0 = blockIdx.y * 64;
  const _Float16* aRow = xh + (size_t)(t0 + ln) * DM + khalf;
  const _Float16* bCol = WT + (size_t)(n0 + ln) * DM + khalf;
  v8f acc[4] = {};
  gemm_16x64(aRow, bCol, DM, DM, acc);
#pragma unroll
  for (int s = 0; s < 4; ++s) {
    int col = n0 + s * 16 + ln;
    float bv = bias[col];
    int h = col >> 6, dh = col & 63;
#pragma unroll
    for (int r = 0; r < 8; ++r) {
      int t = t0 + r + 8 * hl;
      int b = t >> 11, sp = t & (SS - 1);
      float v = acc[s][r] + bv;
      size_t off = (mode == 0)
                       ? ((((size_t)b * HN + h) * SS + sp) * DHD + dh)
                       : ((((size_t)b * HN + h) * DHD + dh) * SS + sp);
      dst[off] = (_Float16)v;
    }
  }
}

// ---------------- flash attention: one wave per (b,h,qtile16) ----------------
__global__ __launch_bounds__(32) void k_attn(const _Float16* __restrict__ qg,
                                             const _Float16* __restrict__ kg,
                                             const _Float16* __restrict__ vTg,
                                             _Float16* __restrict__ outp) {
  int bid = blockIdx.x;
  int bh = bid >> 7;          // S/16 = 128 q tiles
  int qt = bid & 127;
  int lane = threadIdx.x, ln = lane & 15, hl = lane >> 4, khalf = 8 * hl;
  const _Float16* qb = qg + (size_t)bh * SS * DHD;
  const _Float16* kb = kg + (size_t)bh * SS * DHD;
  const _Float16* vb = vTg + (size_t)bh * DHD * SS;
  int q0 = qt * 16;
  v16h aq0 = frag16(qb + (size_t)(q0 + ln) * DHD + khalf);
  v16h aq1 = frag16(qb + (size_t)(q0 + ln) * DHD + khalf + 32);

  float Mrow[8], Spart[8];
  v8f oacc[4] = {};
#pragma unroll
  for (int r = 0; r < 8; ++r) { Mrow[r] = -1e30f; Spart[r] = 0.0f; }

  __shared__ __align__(16) _Float16 P[16 * 32];

  for (int k0 = 0; k0 < SS; k0 += 32) {
    v8f sc[2] = {};
#pragma unroll
    for (int j = 0; j < 2; ++j) {
      const _Float16* bp = kb + (size_t)(k0 + j * 16 + ln) * DHD + khalf;
      sc[j] = wmma_f16(aq0, frag16(bp), sc[j]);
      sc[j] = wmma_f16(aq1, frag16(bp + 32), sc[j]);
    }
#pragma unroll
    for (int r = 0; r < 8; ++r) {
      float s0 = sc[0][r] * 0.125f, s1 = sc[1][r] * 0.125f;
      float m = fmaxf(s0, s1);
#pragma unroll
      for (int off = 8; off; off >>= 1) m = fmaxf(m, __shfl_xor(m, off)); // 16-lane row group
      float newM = fmaxf(Mrow[r], m);
      float corr = __expf(Mrow[r] - newM);
      Mrow[r] = newM;
      float p0 = __expf(s0 - newM), p1 = __expf(s1 - newM);
      Spart[r] = Spart[r] * corr + p0 + p1;
#pragma unroll
      for (int s = 0; s < 4; ++s) oacc[s][r] *= corr;
      int prow = r + 8 * hl;
      P[prow * 32 + ln] = (_Float16)p0;
      P[prow * 32 + 16 + ln] = (_Float16)p1;
    }
    asm volatile("s_wait_dscnt 0" ::: "memory");
    v16h ap = frag16(&P[ln * 32 + khalf]);
#pragma unroll
    for (int s = 0; s < 4; ++s) {
      const _Float16* bp = vb + (size_t)(s * 16 + ln) * SS + k0 + khalf;
      oacc[s] = wmma_f16(ap, frag16(bp), oacc[s]);
    }
    asm volatile("s_wait_dscnt 0" ::: "memory");
  }

  int b = bh >> 3, hh = bh & 7;
#pragma unroll
  for (int r = 0; r < 8; ++r) {
    float s = Spart[r];
#pragma unroll
    for (int off = 8; off; off >>= 1) s += __shfl_xor(s, off);
    float inv = 1.0f / s;
#pragma unroll
    for (int sb = 0; sb < 4; ++sb) {
      size_t t = (size_t)b * SS + q0 + r + 8 * hl;
      outp[t * DM + hh * 64 + sb * 16 + ln] = (_Float16)(oacc[sb][r] * inv);
    }
  }
}

// ---------------- output projection + residual (f32 out) ----------------
__global__ __launch_bounds__(32) void k_gemm_out(const _Float16* __restrict__ ah,
                                                 const _Float16* __restrict__ WT,
                                                 const float* __restrict__ bias,
                                                 const float* __restrict__ res,
                                                 float* __restrict__ out) {
  int lane = threadIdx.x, ln = lane & 15, hl = lane >> 4, khalf = 8 * hl;
  int t0 = blockIdx.x * 16, n0 = blockIdx.y * 64;
  const _Float16* aRow = ah + (size_t)(t0 + ln) * DM + khalf;
  const _Float16* bCol = WT + (size_t)(n0 + ln) * DM + khalf;
  v8f acc[4] = {};
  gemm_16x64(aRow, bCol, DM, DM, acc);
#pragma unroll
  for (int s = 0; s < 4; ++s) {
    int col = n0 + s * 16 + ln;
    float bv = bias[col];
#pragma unroll
    for (int r = 0; r < 8; ++r) {
      size_t t = t0 + r + 8 * hl;
      out[t * DM + col] = acc[s][r] + bv + res[t * DM + col];
    }
  }
}

// ---------------- layernorm (block = 256 = one token) ----------------
__global__ void k_ln(const float* __restrict__ in, const float* __restrict__ g,
                     const float* __restrict__ b, float* __restrict__ out,
                     _Float16* __restrict__ outh) {
  int t = blockIdx.x, tid = threadIdx.x;
  const float* row = in + (size_t)t * DM;
  float a0 = row[tid], a1 = row[tid + 256];
  __shared__ float red[8];
  __shared__ float bc[2];
  float ps = a0 + a1;
#pragma unroll
  for (int off = 16; off; off >>= 1) ps += __shfl_xor(ps, off);
  if ((tid & 31) == 0) red[tid >> 5] = ps;
  __syncthreads();
  if (tid == 0) { float s = 0; for (int i = 0; i < 8; ++i) s += red[i]; bc[0] = s * (1.0f / DM); }
  __syncthreads();
  float mu = bc[0];
  float d0 = a0 - mu, d1 = a1 - mu;
  float pv = d0 * d0 + d1 * d1;
#pragma unroll
  for (int off = 16; off; off >>= 1) pv += __shfl_xor(pv, off);
  if ((tid & 31) == 0) red[tid >> 5] = pv;
  __syncthreads();
  if (tid == 0) { float s = 0; for (int i = 0; i < 8; ++i) s += red[i]; bc[1] = rsqrtf(s * (1.0f / DM) + 1e-6f); }
  __syncthreads();
  float rs = bc[1];
  float o0 = d0 * rs * g[tid] + b[tid];
  float o1 = d1 * rs * g[tid + 256] + b[tid + 256];
  size_t base = (size_t)t * DM;
  out[base + tid] = o0; out[base + tid + 256] = o1;
  outh[base + tid] = (_Float16)o0; outh[base + tid + 256] = (_Float16)o1;
}

// ---------------- MoE gating: softmax(xWg+bg), top-2, renorm, count ----------------
__global__ void k_gate(const float* __restrict__ x, const float* __restrict__ Wg,
                       const float* __restrict__ bg, int* __restrict__ sel,
                       float* __restrict__ selw, int* __restrict__ counts) {
  int t = blockIdx.x, tid = threadIdx.x;
  int w = tid >> 5, ln = tid & 31;
  const float* row = x + (size_t)t * DM;
  float p = 0.0f;
  for (int d = ln; d < DM; d += 32) p += row[d] * Wg[d * EN + w];
#pragma unroll
  for (int off = 16; off; off >>= 1) p += __shfl_xor(p, off);
  __shared__ float logit[EN];
  if (ln == 0) logit[w] = p + bg[w];
  __syncthreads();
  if (tid == 0) {
    float m = logit[0];
    for (int e = 1; e < EN; ++e) m = fmaxf(m, logit[e]);
    float pr[EN], s = 0.0f;
    for (int e = 0; e < EN; ++e) { pr[e] = __expf(logit[e] - m); s += pr[e]; }
    float invs = 1.0f / s;
    for (int e = 0; e < EN; ++e) pr[e] *= invs;
    int i0 = 0;
    for (int e = 1; e < EN; ++e) if (pr[e] > pr[i0]) i0 = e;
    int i1 = (i0 == 0) ? 1 : 0;
    for (int e = 0; e < EN; ++e) if (e != i0 && pr[e] > pr[i1]) i1 = e;
    float v0 = pr[i0], v1 = pr[i1], inv2 = 1.0f / (v0 + v1);
    sel[t * 2] = i0; sel[t * 2 + 1] = i1;
    selw[t * 2] = v0 * inv2; selw[t * 2 + 1] = v1 * inv2;
    atomicAdd(&counts[i0], 1);
    atomicAdd(&counts[i1], 1);
  }
}

__global__ void k_zero8(int* p) { if (threadIdx.x < EN) p[threadIdx.x] = 0; }

__global__ void k_offsets(const int* __restrict__ counts, int* __restrict__ offs,
                          int* __restrict__ cursor, int* __restrict__ ends) {
  if (threadIdx.x == 0 && blockIdx.x == 0) {
    int run = 0;
    for (int e = 0; e < EN; ++e) {
      offs[e] = run; cursor[e] = run; ends[e] = run + counts[e];
      run += (counts[e] + 15) & ~15;   // pad segments to 16 so tiles never straddle experts
    }
  }
}

__global__ void k_scatter(const int* __restrict__ sel, const float* __restrict__ selw,
                          int* __restrict__ cursor, int* __restrict__ slotTok,
                          float* __restrict__ slotW, int* __restrict__ slotOf) {
  int i = blockIdx.x * 256 + threadIdx.x;
  if (i >= TT * 2) return;
  int e = sel[i];
  int slot = atomicAdd(&cursor[e], 1);
  slotTok[slot] = i >> 1;
  slotW[slot] = selw[i];
  slotOf[i] = slot;
}

// ---------------- expert FFN layer 1: h = relu(x W1[e] + b1[e]) ----------------
__global__ __launch_bounds__(32) void k_moe1(const _Float16* __restrict__ xh,
                                             const _Float16* __restrict__ W1T,
                                             const float* __restrict__ b1,
                                             const int* __restrict__ offs,
                                             const int* __restrict__ ends,
                                             const int* __restrict__ slotTok,
                                             _Float16* __restrict__ hbuf) {
  int e = blockIdx.z;
  int tile0 = offs[e] + blockIdx.x * 16;
  int end = ends[e];
  if (tile0 >= end) return;
  int lane = threadIdx.x, ln = lane & 15, hl = lane >> 4, khalf = 8 * hl;
  int n0 = blockIdx.y * 64;
  int slotA = tile0 + ln;
  int tok = slotTok[slotA < end ? slotA : tile0];
  const _Float16* aRow = xh + (size_t)tok * DM + khalf;
  const _Float16* bCol = W1T + ((size_t)e * FF + n0 + ln) * DM + khalf;
  v8f acc[4] = {};
  gemm_16x64(aRow, bCol, DM, DM, acc);
#pragma unroll
  for (int s = 0; s < 4; ++s) {
    int f = n0 + s * 16 + ln;
    float bv = b1[e * FF + f];
#pragma unroll
    for (int r = 0; r < 8; ++r) {
      int slot = tile0 + r + 8 * hl;
      if (slot < end) hbuf[(size_t)slot * FF + f] = (_Float16)fmaxf(acc[s][r] + bv, 0.0f);
    }
  }
}

// ---------------- expert FFN layer 2: o = h W2[e] + b2[e] (per slot) ----------------
__global__ __launch_bounds__(32) void k_moe2(const _Float16* __restrict__ hbuf,
                                             const _Float16* __restrict__ W2T,
                                             const float* __restrict__ b2,
                                             const int* __restrict__ offs,
                                             const int* __restrict__ ends,
                                             float* __restrict__ obuf) {
  int e = blockIdx.z;
  int tile0 = offs[e] + blockIdx.x * 16;
  int end = ends[e];
  if (tile0 >= end) return;
  int lane = threadIdx.x, ln = lane & 15, hl = lane >> 4, khalf = 8 * hl;
  int n0 = blockIdx.y * 64;
  int slotA = tile0 + ln;
  int sA = slotA < end ? slotA : tile0;
  const _Float16* aRow = hbuf + (size_t)sA * FF + khalf;
  const _Float16* bCol = W2T + ((size_t)e * DM + n0 + ln) * FF + khalf;
  v8f acc[4] = {};
  gemm_16x64(aRow, bCol, FF, FF, acc);
#pragma unroll
  for (int s = 0; s < 4; ++s) {
    int d = n0 + s * 16 + ln;
    float bv = b2[e * DM + d];
#pragma unroll
    for (int r = 0; r < 8; ++r) {
      int slot = tile0 + r + 8 * hl;
      if (slot < end) obuf[(size_t)slot * DM + d] = acc[s][r] + bv;
    }
  }
}

// deterministic combine: y = out1 + gw0*o[slot0] + gw1*o[slot1]
__global__ void k_combine(const float* __restrict__ out1, const float* __restrict__ obuf,
                          const int* __restrict__ slotOf, const float* __restrict__ selw,
                          float* __restrict__ y) {
  int idx = blockIdx.x * 256 + threadIdx.x;
  if (idx >= TT * DM) return;
  int t = idx >> 9, d = idx & 511;
  float v = out1[idx];
  v += selw[t * 2]     * obuf[(size_t)slotOf[t * 2] * DM + d];
  v += selw[t * 2 + 1] * obuf[(size_t)slotOf[t * 2 + 1] * DM + d];
  y[idx] = v;
}

// ---------------- host orchestration ----------------
extern "C" void kernel_launch(void* const* d_in, const int* in_sizes, int n_in,
                              void* d_out, int out_size, void* d_ws, size_t ws_size,
                              hipStream_t stream) {
  const int*   tokens = (const int*)d_in[0];
  const float* emb = (const float*)d_in[1];
  const float* Wq = (const float*)d_in[2];
  const float* bq = (const float*)d_in[3];
  const float* Wk = (const float*)d_in[4];
  const float* bk = (const float*)d_in[5];
  const float* Wv = (const float*)d_in[6];
  const float* bv = (const float*)d_in[7];
  const float* Wo = (const float*)d_in[8];
  const float* bo = (const float*)d_in[9];
  const float* Wg = (const float*)d_in[10];
  const float* bg = (const float*)d_in[11];
  const float* W1 = (const float*)d_in[12];
  const float* b1 = (const float*)d_in[13];
  const float* W2 = (const float*)d_in[14];
  const float* b2 = (const float*)d_in[15];
  const float* g1 = (const float*)d_in[16];
  const float* be1 = (const float*)d_in[17];
  const float* g2 = (const float*)d_in[18];
  const float* be2 = (const float*)d_in[19];

  uint8_t* base = (uint8_t*)d_ws;
  size_t off = 0;
  auto carve = [&](size_t bytes) -> void* {
    void* r = base + off;
    off += (bytes + 255) & ~(size_t)255;
    return r;
  };
  const size_t TD = (size_t)TT * DM;
  float*    x       = (float*)   carve(TD * 4);
  _Float16* xh      = (_Float16*)carve(TD * 2);
  _Float16* qh      = (_Float16*)carve(TD * 2);
  _Float16* kh      = (_Float16*)carve(TD * 2);
  _Float16* vTh     = (_Float16*)carve(TD * 2);
  _Float16* attnh   = (_Float16*)carve(TD * 2);
  float*    attnsum = (float*)   carve(TD * 4);
  float*    out1    = (float*)   carve(TD * 4);
  _Float16* out1h   = (_Float16*)carve(TD * 2);
  float*    y       = (float*)   carve(TD * 4);
  _Float16* WTq     = (_Float16*)carve((size_t)DM * DM * 2);
  _Float16* WTk     = (_Float16*)carve((size_t)DM * DM * 2);
  _Float16* WTv     = (_Float16*)carve((size_t)DM * DM * 2);
  _Float16* WTo     = (_Float16*)carve((size_t)DM * DM * 2);
  _Float16* W1T     = (_Float16*)carve((size_t)EN * DM * FF * 2);
  _Float16* W2T     = (_Float16*)carve((size_t)EN * DM * FF * 2);
  int*      counts  = (int*)     carve(EN * 4);
  int*      offsA   = (int*)     carve(EN * 4);
  int*      cursor  = (int*)     carve(EN * 4);
  int*      endsA   = (int*)     carve(EN * 4);
  int*      sel     = (int*)     carve((size_t)TT * 2 * 4);
  float*    selw    = (float*)   carve((size_t)TT * 2 * 4);
  int*      slotTok = (int*)     carve((size_t)SLOTCAP * 4);
  float*    slotW   = (float*)   carve((size_t)SLOTCAP * 4);
  int*      slotOf  = (int*)     carve((size_t)TT * 2 * 4);
  float*    obuf    = (float*)   carve((size_t)SLOTCAP * DM * 4);
  _Float16* hbuf    = (_Float16*)carve((size_t)SLOTCAP * FF * 2);
  if (off > ws_size) return;  // workspace too small — bail rather than corrupt

  k_embed<<<(TT * DM) / 256, 256, 0, stream>>>(tokens, emb, x, xh);

  for (int l = 0; l < LNUM; ++l) {
    const size_t wOff = (size_t)l * DM * DM;
    // weight transpose-convert to f16 (B operand wants [N][K] contiguous rows)
    k_convT<<<(DM * DM) / 256, 256, 0, stream>>>(Wq + wOff, WTq, DM, DM, DM * DM);
    k_convT<<<(DM * DM) / 256, 256, 0, stream>>>(Wk + wOff, WTk, DM, DM, DM * DM);
    k_convT<<<(DM * DM) / 256, 256, 0, stream>>>(Wv + wOff, WTv, DM, DM, DM * DM);
    k_convT<<<(DM * DM) / 256, 256, 0, stream>>>(Wo + wOff, WTo, DM, DM, DM * DM);

    dim3 gProj(TT / 16, DM / 64);
    k_gemm_qkv<<<gProj, 32, 0, stream>>>(xh, WTq, bq + l * DM, qh, 0);
    k_gemm_qkv<<<gProj, 32, 0, stream>>>(xh, WTk, bk + l * DM, kh, 0);
    k_gemm_qkv<<<gProj, 32, 0, stream>>>(xh, WTv, bv + l * DM, vTh, 1);

    k_attn<<<BB * HN * (SS / 16), 32, 0, stream>>>(qh, kh, vTh, attnh);

    k_gemm_out<<<gProj, 32, 0, stream>>>(attnh, WTo, bo + l * DM, x, attnsum);
    k_ln<<<TT, 256, 0, stream>>>(attnsum, g1 + l * DM, be1 + l * DM, out1, out1h);

    // MoE: gate -> route -> expert GEMMs -> combine
    k_zero8<<<1, 32, 0, stream>>>(counts);
    k_gate<<<TT, 256, 0, stream>>>(out1, Wg + (size_t)l * DM * EN, bg + l * EN,
                                   sel, selw, counts);
    k_offsets<<<1, 1, 0, stream>>>(counts, offsA, cursor, endsA);
    k_scatter<<<(TT * 2) / 256, 256, 0, stream>>>(sel, selw, cursor, slotTok, slotW, slotOf);

    k_convT<<<(EN * DM * FF) / 256, 256, 0, stream>>>(W1 + (size_t)l * EN * DM * FF, W1T,
                                                      DM, FF, EN * DM * FF);
    k_convT<<<(EN * DM * FF) / 256, 256, 0, stream>>>(W2 + (size_t)l * EN * DM * FF, W2T,
                                                      FF, DM, EN * DM * FF);

    k_moe1<<<dim3(TT / 16, FF / 64, EN), 32, 0, stream>>>(out1h, W1T, b1 + l * EN * FF,
                                                          offsA, endsA, slotTok, hbuf);
    k_moe2<<<dim3(TT / 16, DM / 64, EN), 32, 0, stream>>>(hbuf, W2T, b2 + l * EN * DM,
                                                          offsA, endsA, obuf);
    k_combine<<<(TT * DM) / 256, 256, 0, stream>>>(out1, obuf, slotOf, selw, y);

    float* xNext = (l == LNUM - 1) ? (float*)d_out : x;
    k_ln<<<TT, 256, 0, stream>>>(y, g2 + l * DM, be2 + l * DM, xNext, xh);
  }
}
